// FFTAttention_21431886807686
// MI455X (gfx1250) — matrix operand order repaired and checked by reading, not compiled, not run on hardware
//
#include <hip/hip_runtime.h>
#include <hip/hip_bf16.h>

typedef __attribute__((ext_vector_type(16))) _Float16 v16h;
typedef __attribute__((ext_vector_type(8)))  _Float16 v8h;
typedef __attribute__((ext_vector_type(8)))  float    v8f;
typedef __attribute__((ext_vector_type(4)))  unsigned int u32x4;
typedef __attribute__((ext_vector_type(4)))  int          i32x4;
typedef __attribute__((ext_vector_type(8)))  int          i32x8;

#define PI_F 3.14159265358979323846f

#if __has_include(<hip/amd_detail/amd_gfx1250_TDM.h>)
#define TDM_SIX_ARGS 1
#endif

// ---------------------------------------------------------------------------
// Tensor Data Mover: 2-D f16 tile load, global -> LDS, with LDS row padding.
// pad_int_code: 2^(code+1) dwords between pads; pad_amt_code: (code+1) dwords.
// ---------------------------------------------------------------------------
__device__ __forceinline__ void tdm_load_2d_f16(const _Float16* gsrc, void* lds_dst,
                                                unsigned tile_x, unsigned tile_y,
                                                unsigned tensor_x, unsigned tensor_y,
                                                unsigned stride_x,
                                                unsigned pad_int_code,
                                                unsigned pad_amt_code) {
    unsigned long long ga = (unsigned long long)(uintptr_t)gsrc;
    unsigned lds = (unsigned)(uintptr_t)lds_dst;   // low 32 bits = LDS byte offset
    u32x4 g0;
    g0[0] = 1u;                                        // count=1 (valid), user mode
    g0[1] = lds;                                       // lds_addr
    g0[2] = (unsigned)ga;                              // global_addr[31:0]
    g0[3] = (unsigned)((ga >> 32) & 0x01ffffffu) | (2u << 30);  // addr[56:32] | type=2
    i32x8 g1;
    g1[0] = (int)((1u << 16) |                          // data_size = 2 bytes
                  (1u << 20) |                          // pad_enable
                  (pad_int_code << 22) | (pad_amt_code << 25));
    g1[1] = (int)((tensor_x & 0xffffu) << 16);          // tensor_dim0[15:0]
    g1[2] = (int)(((tensor_x >> 16) & 0xffffu) | ((tensor_y & 0xffffu) << 16));
    g1[3] = (int)(((tensor_y >> 16) & 0xffffu) | ((tile_x & 0xffffu) << 16));
    g1[4] = (int)(tile_y & 0xffffu);                    // tile_dim1; tile_dim2=0 -> 2D
    g1[5] = (int)stride_x;                              // tensor_dim0_stride[31:0]
    g1[6] = 0;
    g1[7] = 0;
    i32x4 z4 = {};
#ifdef TDM_SIX_ARGS
    i32x8 z8 = {};
    __builtin_amdgcn_tensor_load_to_lds(g0, g1, z4, z4, z8, 0);
#else
    __builtin_amdgcn_tensor_load_to_lds(g0, g1, z4, z4, 0);
#endif
}

// ---------------------------------------------------------------------------
// WMMA fragment helpers (CDNA5 16x16x32 f16, wave32)
// ---------------------------------------------------------------------------
__device__ __forceinline__ v16h load_frag_a(const _Float16* tile, int stride) {
    const int lane = threadIdx.x & 31;
    const int m  = lane & 15;
    const int hh = lane >> 4;
    const _Float16* p = tile + m * stride + hh * 8;
    v8h lo = *(const v8h*)(p);
    v8h hi = *(const v8h*)(p + 16);
    v16h r;
#pragma unroll
    for (int j = 0; j < 8; ++j) { r[j] = lo[j]; r[j + 8] = hi[j]; }
    return r;
}

__device__ __forceinline__ v16h load_frag_b(const _Float16* tile, int stride) {
    const int lane = threadIdx.x & 31;
    const int n  = lane & 15;
    const int hh = lane >> 4;
    const _Float16* p = tile + n * stride + hh * 8;
    v8h lo = *(const v8h*)(p);
    v8h hi = *(const v8h*)(p + 16);
    v16h r;
#pragma unroll
    for (int j = 0; j < 8; ++j) { r[j] = lo[j]; r[j + 8] = hi[j]; }
    return r;
}

__device__ __forceinline__ v8f wmma16(v16h a, v16h b, v8f c) {
    return __builtin_amdgcn_wmma_f32_16x16x32_f16(false, a, false, b,
                                                  (short)0, c, false, false);
}

// ---------------------------------------------------------------------------
// Kernel 0: one-time f32 -> f16 weight conversion (makes A tiles TDM-able).
// ---------------------------------------------------------------------------
__global__ void __launch_bounds__(256) cvt_f16(const float* __restrict__ in,
                                               _Float16* __restrict__ out, int n) {
    int i = blockIdx.x * 256 + threadIdx.x;
    if (i < n) out[i] = (_Float16)in[i];
}

// ---------------------------------------------------------------------------
// Kernel 1: forward FFT2 (32x32) as two 32-point DFT passes per image.
// Input staged with GLOBAL_LOAD_ASYNC_TO_LDS (ASYNCcnt path).
// ---------------------------------------------------------------------------
__global__ void __launch_bounds__(256) fft2_fwd(const float* __restrict__ x,
                                                float* __restrict__ xr) {
    const int img = blockIdx.x;  // b*256 + c
    __shared__ __align__(16) float Xs[32 * 33];
    __shared__ __align__(16) float Tr[32 * 33], Ti[32 * 33];
    const float* xp = x + (size_t)img * 1024;
#pragma unroll
    for (int it = 0; it < 4; ++it) {
        int i = threadIdx.x + it * 256;
        unsigned lo = (unsigned)(uintptr_t)&Xs[(i >> 5) * 33 + (i & 31)];
        unsigned go = (unsigned)(i * 4);
        asm volatile("global_load_async_to_lds_b32 %0, %1, %2"
                     :: "v"(lo), "v"(go), "s"(xp) : "memory");
    }
    asm volatile("s_wait_asynccnt 0" ::: "memory");
    __syncthreads();
    for (int i = threadIdx.x; i < 1024; i += 256) {
        int r = i >> 5, k = i & 31;
        float sr = 0.f, si = 0.f;
        for (int n = 0; n < 32; ++n) {
            float ang = -(2.0f * PI_F / 32.0f) * (float)((k * n) & 31);
            float sn, cs; __sincosf(ang, &sn, &cs);
            float xv = Xs[r * 33 + n];
            sr += xv * cs;
            si += xv * sn;
        }
        Tr[r * 33 + k] = sr;
        Ti[r * 33 + k] = si;
    }
    __syncthreads();
    float* op = xr + (size_t)img * 2048;
    for (int i = threadIdx.x; i < 1024; i += 256) {
        int k1 = i >> 5, k2 = i & 31;
        float sr = 0.f, si = 0.f;
        for (int r = 0; r < 32; ++r) {
            float ang = -(2.0f * PI_F / 32.0f) * (float)((k1 * r) & 31);
            float sn, cs; __sincosf(ang, &sn, &cs);
            float tr = Tr[r * 33 + k2], ti = Ti[r * 33 + k2];
            sr += tr * cs - ti * sn;
            si += tr * sn + ti * cs;
        }
        op[2 * i]     = sr;
        op[2 * i + 1] = si;
    }
}

// ---------------------------------------------------------------------------
// Kernel 2: GroupNorm; writes normalized tensor transposed: n16T[b][n][c] f16.
// ---------------------------------------------------------------------------
__global__ void __launch_bounds__(256) groupnorm(const float* __restrict__ xr,
                                                 const float* __restrict__ gw,
                                                 const float* __restrict__ gb,
                                                 _Float16* __restrict__ n16T) {
    const int b = blockIdx.x >> 5;
    const int g = blockIdx.x & 31;
    const int tid = threadIdx.x;
    const float* base = xr + ((size_t)b * 256 + (size_t)g * 8) * 2048;
    float s = 0.f, s2 = 0.f;
    for (int i = tid; i < 16384; i += 256) {
        float v = base[i];
        s += v; s2 += v * v;
    }
    __shared__ float rs[256], rq[256];
    rs[tid] = s; rq[tid] = s2;
    __syncthreads();
    for (int off = 128; off > 0; off >>= 1) {
        if (tid < off) { rs[tid] += rs[tid + off]; rq[tid] += rq[tid + off]; }
        __syncthreads();
    }
    float mean = rs[0] * (1.0f / 16384.0f);
    float var  = rq[0] * (1.0f / 16384.0f) - mean * mean;
    float rstd = rsqrtf(var + 1e-5f);
    for (int i = tid; i < 16384; i += 256) {
        int cl = i >> 11;
        int n  = i & 2047;
        int c  = g * 8 + cl;
        float v = (base[i] - mean) * rstd * gw[c] + gb[c];
        n16T[((size_t)b * 2048 + n) * 256 + c] = (_Float16)v;
    }
}

// ---------------------------------------------------------------------------
// Kernel 3: QKV GEMM (per b: 768x2048 = W(768x256) x N(256x2048)).
// Both A and B tiles DMA'd by TDM, double-buffered (wait tensorcnt<=2 keeps the
// just-issued pair in flight while guaranteeing the current pair landed).
// ---------------------------------------------------------------------------
__global__ void __launch_bounds__(256) qkv_gemm(const _Float16* __restrict__ w16,
                                                const float* __restrict__ bias,
                                                const _Float16* __restrict__ n16T,
                                                _Float16* __restrict__ qT,
                                                _Float16* __restrict__ kT,
                                                _Float16* __restrict__ vv) {
    const int b  = blockIdx.z;
    const int n0 = blockIdx.x * 64;
    const int o0 = blockIdx.y * 64;
    const int tid = threadIdx.x, wave = tid >> 5, lane = tid & 31;
    const int hh = lane >> 4, ln = lane & 15;
    const int wm = (wave & 3) * 16;
    const int wn = (wave >> 2) * 32;
    __shared__ __align__(16) _Float16 As[2][64 * 40];  // [o][k]
    __shared__ __align__(16) _Float16 Bs[2][64 * 40];  // [n][k]
    const _Float16* wp = w16 + (size_t)o0 * 256;
    const _Float16* np = n16T + ((size_t)b * 2048 + n0) * 256;
    if (wave == 0) {
        tdm_load_2d_f16(wp, As[0], 32, 64, 32, 64, 256, 3, 3);
        tdm_load_2d_f16(np, Bs[0], 32, 64, 32, 64, 256, 3, 3);
    }
    v8f acc0 = {}, acc1 = {};
    for (int k0 = 0; k0 < 256; k0 += 32) {
        const int buf = (k0 >> 5) & 1;
        if (wave == 0) {
            if (k0 + 32 < 256) {
                tdm_load_2d_f16(wp + k0 + 32, As[buf ^ 1], 32, 64, 32, 64, 256, 3, 3);
                tdm_load_2d_f16(np + k0 + 32, Bs[buf ^ 1], 32, 64, 32, 64, 256, 3, 3);
                __builtin_amdgcn_s_wait_tensorcnt(2);
            } else {
                __builtin_amdgcn_s_wait_tensorcnt(0);
            }
        }
        __syncthreads();
        v16h a  = load_frag_a(As[buf] + wm * 40, 40);
        v16h b0 = load_frag_b(Bs[buf] + wn * 40, 40);
        v16h b1 = load_frag_b(Bs[buf] + (wn + 16) * 40, 40);
        acc0 = wmma16(a, b0, acc0);
        acc1 = wmma16(a, b1, acc1);
        __syncthreads();
    }
    const float scale = 0.35355339059327373f;  // 64^-0.25
    v8f accs[2] = {acc0, acc1};
#pragma unroll
    for (int half = 0; half < 2; ++half) {
#pragma unroll
        for (int r = 0; r < 8; ++r) {
            int o = o0 + wm + r + 8 * hh;
            int n = n0 + wn + half * 16 + ln;
            float val = accs[half][r] + bias[o];
            int hd = o / 192, rr = o % 192;
            size_t bh = (size_t)b * 4 + hd;
            if (rr < 64)
                qT[(bh * 2048 + n) * 64 + rr] = (_Float16)(val * scale);
            else if (rr < 128)
                kT[(bh * 2048 + n) * 64 + (rr - 64)] = (_Float16)(val * scale);
            else
                vv[(bh * 64 + (rr - 128)) * 2048 + n] = (_Float16)val;
        }
    }
}

// ---------------------------------------------------------------------------
// Kernel 4: flash attention. K/V tiles double-buffered through the TDM.
// ---------------------------------------------------------------------------
__global__ void __launch_bounds__(256) flash_attn(const _Float16* __restrict__ qT,
                                                  const _Float16* __restrict__ kT,
                                                  const _Float16* __restrict__ vv,
                                                  _Float16* __restrict__ aT) {
    const int bh = blockIdx.y;
    const int b  = bh >> 2;
    const int h  = bh & 3;
    const int t0 = blockIdx.x * 128;
    const int tid  = threadIdx.x;
    const int wave = tid >> 5;
    const int lane = tid & 31;
    const int hh = lane >> 4, ln = lane & 15;

    __shared__ __align__(16) _Float16 Qs[128 * 72];      // [t][c]
    __shared__ __align__(16) _Float16 Ks[2][64 * 72];    // [s][c]
    __shared__ __align__(16) _Float16 Vs[2][64 * 72];    // [c][s]
    __shared__ __align__(16) _Float16 Ps[128 * 72];      // [t][s]

    const _Float16* kbase = kT + (size_t)bh * 2048 * 64;
    const _Float16* vbase = vv + (size_t)bh * 64 * 2048;

    if (wave == 0) {
        tdm_load_2d_f16(qT + ((size_t)bh * 2048 + t0) * 64, Qs,
                        64, 128, 64, 2048, 64, 4, 3);              // Q strip
        tdm_load_2d_f16(kbase, Ks[0], 64, 64, 64, 2048, 64, 4, 3); // K s-block 0
        tdm_load_2d_f16(vbase, Vs[0], 64, 64, 2048, 64, 2048, 4, 3);
    }

    float mrow[8], lrow[8];
#pragma unroll
    for (int r = 0; r < 8; ++r) { mrow[r] = -3.0e38f; lrow[r] = 0.0f; }
    v8f acc[4] = {};

    for (int s0 = 0; s0 < 2048; s0 += 64) {
        const int buf = (s0 >> 6) & 1;
        if (wave == 0) {
            if (s0 + 64 < 2048) {   // prefetch next s-block into the other buffer
                tdm_load_2d_f16(kbase + (size_t)(s0 + 64) * 64, Ks[buf ^ 1],
                                64, 64, 64, 2048, 64, 4, 3);
                tdm_load_2d_f16(vbase + s0 + 64, Vs[buf ^ 1],
                                64, 64, 2048, 64, 2048, 4, 3);
                __builtin_amdgcn_s_wait_tensorcnt(2);  // current pair (+Q) landed
            } else {
                __builtin_amdgcn_s_wait_tensorcnt(0);
            }
        }
        __syncthreads();

        // S strip (16 x 64) = Q strip x K^T block, K-dim = c = 64
        const _Float16* qrow = Qs + wave * 16 * 72;
        v16h aq0 = load_frag_a(qrow, 72);
        v16h aq1 = load_frag_a(qrow + 32, 72);
        v8f sfr[4];
#pragma unroll
        for (int js = 0; js < 4; ++js) {
            v8f t = {};
            t = wmma16(aq0, load_frag_b(Ks[buf] + js * 16 * 72, 72), t);
            t = wmma16(aq1, load_frag_b(Ks[buf] + js * 16 * 72 + 32, 72), t);
            sfr[js] = t;
        }

        // online softmax (rows live in 16-lane halves)
#pragma unroll
        for (int r = 0; r < 8; ++r) {
            float mx = fmaxf(fmaxf(sfr[0][r], sfr[1][r]),
                             fmaxf(sfr[2][r], sfr[3][r]));
#pragma unroll
            for (int off = 1; off < 16; off <<= 1)
                mx = fmaxf(mx, __shfl_xor(mx, off, 32));
            float nm = fmaxf(mrow[r], mx);
            float alpha = __expf(mrow[r] - nm);
            float rsum = 0.f;
#pragma unroll
            for (int js = 0; js < 4; ++js) {
                float p = __expf(sfr[js][r] - nm);
                sfr[js][r] = p;
                rsum += p;
            }
#pragma unroll
            for (int off = 1; off < 16; off <<= 1)
                rsum += __shfl_xor(rsum, off, 32);
            lrow[r] = lrow[r] * alpha + rsum;
            mrow[r] = nm;
#pragma unroll
            for (int jc = 0; jc < 4; ++jc) acc[jc][r] = acc[jc][r] * alpha;
        }

        // write P (D-layout) to LDS to reshape as an A fragment
#pragma unroll
        for (int r = 0; r < 8; ++r) {
            int trow = wave * 16 + r + 8 * hh;
#pragma unroll
            for (int js = 0; js < 4; ++js)
                Ps[trow * 72 + js * 16 + ln] = (_Float16)sfr[js][r];
        }
        __syncthreads();

        // O strip += P (16x64) x V^T block (64x64)
        const _Float16* prow = Ps + wave * 16 * 72;
        v16h ap0 = load_frag_a(prow, 72);
        v16h ap1 = load_frag_a(prow + 32, 72);
#pragma unroll
        for (int jc = 0; jc < 4; ++jc) {
            acc[jc] = wmma16(ap0, load_frag_b(Vs[buf] + jc * 16 * 72, 72), acc[jc]);
            acc[jc] = wmma16(ap1, load_frag_b(Vs[buf] + jc * 16 * 72 + 32, 72), acc[jc]);
        }
        __syncthreads();
    }

    // epilogue -> aT[b][t][256] (transposed for proj B-tiles)
#pragma unroll
    for (int r = 0; r < 8; ++r) {
        int t = t0 + wave * 16 + r + 8 * hh;
        float inv = 1.0f / lrow[r];
#pragma unroll
        for (int jc = 0; jc < 4; ++jc) {
            int c = h * 64 + jc * 16 + ln;
            aT[((size_t)b * 2048 + t) * 256 + c] = (_Float16)(acc[jc][r] * inv);
        }
    }
}

// ---------------------------------------------------------------------------
// Kernel 5: proj GEMM (per b: 256x2048 = W(256x256) x A(256x2048)), f32 out.
// A and B tiles via double-buffered TDM.
// ---------------------------------------------------------------------------
__global__ void __launch_bounds__(256) proj_gemm(const _Float16* __restrict__ w16,
                                                 const float* __restrict__ bias,
                                                 const _Float16* __restrict__ aT,
                                                 float* __restrict__ y) {
    const int b  = blockIdx.z;
    const int n0 = blockIdx.x * 64;
    const int o0 = blockIdx.y * 64;
    const int tid = threadIdx.x, wave = tid >> 5, lane = tid & 31;
    const int hh = lane >> 4, ln = lane & 15;
    const int wm = (wave & 3) * 16;
    const int wn = (wave >> 2) * 32;
    __shared__ __align__(16) _Float16 As[2][64 * 40];
    __shared__ __align__(16) _Float16 Bs[2][64 * 40];
    const _Float16* wp = w16 + (size_t)o0 * 256;
    const _Float16* ap = aT + ((size_t)b * 2048 + n0) * 256;
    if (wave == 0) {
        tdm_load_2d_f16(wp, As[0], 32, 64, 32, 64, 256, 3, 3);
        tdm_load_2d_f16(ap, Bs[0], 32, 64, 32, 64, 256, 3, 3);
    }
    v8f acc0 = {}, acc1 = {};
    for (int k0 = 0; k0 < 256; k0 += 32) {
        const int buf = (k0 >> 5) & 1;
        if (wave == 0) {
            if (k0 + 32 < 256) {
                tdm_load_2d_f16(wp + k0 + 32, As[buf ^ 1], 32, 64, 32, 64, 256, 3, 3);
                tdm_load_2d_f16(ap + k0 + 32, Bs[buf ^ 1], 32, 64, 32, 64, 256, 3, 3);
                __builtin_amdgcn_s_wait_tensorcnt(2);
            } else {
                __builtin_amdgcn_s_wait_tensorcnt(0);
            }
        }
        __syncthreads();
        v16h a  = load_frag_a(As[buf] + wm * 40, 40);
        v16h b0 = load_frag_b(Bs[buf] + wn * 40, 40);
        v16h b1 = load_frag_b(Bs[buf] + (wn + 16) * 40, 40);
        acc0 = wmma16(a, b0, acc0);
        acc1 = wmma16(a, b1, acc1);
        __syncthreads();
    }
    v8f accs[2] = {acc0, acc1};
#pragma unroll
    for (int half = 0; half < 2; ++half) {
#pragma unroll
        for (int r = 0; r < 8; ++r) {
            int o = o0 + wm + r + 8 * hh;
            int n = n0 + wn + half * 16 + ln;
            y[((size_t)b * 256 + o) * 2048 + n] = accs[half][r] + bias[o];
        }
    }
}

// ---------------------------------------------------------------------------
// Kernel 6: inverse FFT2 + stack(real, imag).
// ---------------------------------------------------------------------------
__global__ void __launch_bounds__(256) ifft2_bwd(const float* __restrict__ y,
                                                 float* __restrict__ out) {
    const int img = blockIdx.x;
    __shared__ float Yr[32 * 33], Yi[32 * 33];
    __shared__ float Tr[32 * 33], Ti[32 * 33];
    const float* yp = y + (size_t)img * 2048;
    for (int i = threadIdx.x; i < 1024; i += 256) {
        Yr[(i >> 5) * 33 + (i & 31)] = yp[2 * i];
        Yi[(i >> 5) * 33 + (i & 31)] = yp[2 * i + 1];
    }
    __syncthreads();
    for (int i = threadIdx.x; i < 1024; i += 256) {
        int k1 = i >> 5, m2 = i & 31;
        float sr = 0.f, si = 0.f;
        for (int k2 = 0; k2 < 32; ++k2) {
            float ang = (2.0f * PI_F / 32.0f) * (float)((k2 * m2) & 31);
            float sn, cs; __sincosf(ang, &sn, &cs);
            float yr = Yr[k1 * 33 + k2], yi = Yi[k1 * 33 + k2];
            sr += yr * cs - yi * sn;
            si += yr * sn + yi * cs;
        }
        Tr[k1 * 33 + m2] = sr;
        Ti[k1 * 33 + m2] = si;
    }
    __syncthreads();
    float* op = out + (size_t)img * 2048;
    for (int i = threadIdx.x; i < 1024; i += 256) {
        int m1 = i >> 5, m2 = i & 31;
        float sr = 0.f, si = 0.f;
        for (int k1 = 0; k1 < 32; ++k1) {
            float ang = (2.0f * PI_F / 32.0f) * (float)((k1 * m1) & 31);
            float sn, cs; __sincosf(ang, &sn, &cs);
            float tr = Tr[k1 * 33 + m2], ti = Ti[k1 * 33 + m2];
            sr += tr * cs - ti * sn;
            si += tr * sn + ti * cs;
        }
        op[2 * i]     = sr * (1.0f / 1024.0f);
        op[2 * i + 1] = si * (1.0f / 1024.0f);
    }
}

// ---------------------------------------------------------------------------
extern "C" void kernel_launch(void* const* d_in, const int* in_sizes, int n_in,
                              void* d_out, int out_size, void* d_ws, size_t ws_size,
                              hipStream_t stream) {
    const float* x      = (const float*)d_in[0];
    const float* gn_w   = (const float*)d_in[1];
    const float* gn_b   = (const float*)d_in[2];
    const float* qkv_w  = (const float*)d_in[3];
    const float* qkv_b  = (const float*)d_in[4];
    const float* proj_w = (const float*)d_in[5];
    const float* proj_b = (const float*)d_in[6];
    float* out = (float*)d_out;

    char* ws = (char*)d_ws;
    float*    xr   = (float*)(ws);                      // 16 MB f32 (8,256,2048)
    _Float16* n16T = (_Float16*)(ws + (16ull << 20));   //  8 MB f16 [b][n][256]
    _Float16* qT   = (_Float16*)(ws + (24ull << 20));   //  8 MB f16 [bh][t][64]
    _Float16* kT   = (_Float16*)(ws + (32ull << 20));   //  8 MB f16 [bh][s][64]
    _Float16* vv   = (_Float16*)(ws + (40ull << 20));   //  8 MB f16 [bh][64][s]
    _Float16* aT   = (_Float16*)(ws + (48ull << 20));   //  8 MB f16 [b][n][256]
    float*    y    = (float*)(ws + (56ull << 20));      // 16 MB f32 (8,256,2048)
    _Float16* w16q = (_Float16*)(ws + (72ull << 20));             // 384 KB
    _Float16* w16p = (_Float16*)(ws + (72ull << 20) + (1u << 19)); // 128 KB

    cvt_f16  <<<768, 256, 0, stream>>>(qkv_w, w16q, 768 * 256);
    cvt_f16  <<<256, 256, 0, stream>>>(proj_w, w16p, 256 * 256);
    fft2_fwd <<<2048, 256, 0, stream>>>(x, xr);
    groupnorm<<<256, 256, 0, stream>>>(xr, gn_w, gn_b, n16T);
    qkv_gemm <<<dim3(32, 12, 8), 256, 0, stream>>>(w16q, qkv_b, n16T, qT, kT, vv);
    flash_attn<<<dim3(16, 32), 256, 0, stream>>>(qT, kT, vv, aT);
    proj_gemm<<<dim3(32, 4, 8), 256, 0, stream>>>(w16p, proj_b, aT, y);
    ifft2_bwd<<<2048, 256, 0, stream>>>(y, out);
}
